// HGT_50242527428755
// MI455X (gfx1250) — matrix operand order repaired and checked by reading, not compiled, not run on hardware
//
#include <hip/hip_runtime.h>
#include <math.h>

// ---------------- problem constants ----------------
#define T_      8
#define N_      20000
#define FIN_    128
#define H_      128
#define HEADS_  4
#define D_      32      // H_/HEADS_
#define R_      12
#define E_      40000
#define L_      2
#define C_      16

#define MTILES_  (N_ / 16)          // 1250
#define WPB_     8                  // waves per block in strip kernels
#define MBLKS_   ((MTILES_ + WPB_ - 1) / WPB_)   // 157

typedef __attribute__((ext_vector_type(16))) _Float16 v16h;
typedef __attribute__((ext_vector_type(8)))  float    v8f;

enum { ACT_NONE = 0, ACT_RELU = 1, ACT_GELU = 2 };

__device__ __forceinline__ float gelu_exact(float x) {
    return 0.5f * x * (1.0f + erff(x * 0.70710678118654752f));
}

// IEEE float atomic max via int/uint trick (mixed signs OK; init -1e30f).
__device__ __forceinline__ void atomicMaxFloat(float* addr, float val) {
    if (val >= 0.0f) atomicMax((int*)addr, __float_as_int(val));
    else             atomicMin((unsigned int*)addr, (unsigned int)__float_as_int(val));
}

// ---------------- WMMA fragment loaders -----------------------------------
// A-matrix 16x32 f16 fragment, contiguous k order (same as B):
//   lanes 0-15 : row M=lane,    K = {k0..k0+15} in v[0..15]
//   lanes16-31 : row M=lane-16, K = {k0+16..k0+31}
template<int ACTIN>
__device__ __forceinline__ v16h load_a_frag(const float* __restrict__ A,
                                            int lda, int row0, int k0) {
    const int lane = threadIdx.x & 31;
    const int m  = lane & 15;
    const int hi = lane >> 4;
    const float* p = A + (size_t)(row0 + m) * lda + k0 + hi * 16;
    v16h a;
#pragma unroll
    for (int i = 0; i < 8; ++i) {
        float f = p[i];
        if (ACTIN == ACT_GELU) f = gelu_exact(f);
        a[i] = (_Float16)f;
    }
#pragma unroll
    for (int i = 0; i < 8; ++i) {
        float f = p[8 + i];
        if (ACTIN == ACT_GELU) f = gelu_exact(f);
        a[8 + i] = (_Float16)f;
    }
    return a;
}

// B-matrix 32x16 f16 layout (from f32 row-major weight):
//   lanes 0-15 : col N=lane,    VGPR j holds K = k0+2j, k0+2j+1
//   lanes16-31 : col N=lane-16, VGPR j holds K = k0+16+2j, k0+16+2j+1
__device__ __forceinline__ v16h load_b_frag(const float* __restrict__ W,
                                            int ldw, int k0, int col0) {
    const int lane = threadIdx.x & 31;
    const int n  = lane & 15;
    const int kg = lane >> 4;
    const float* p = W + (size_t)(k0 + kg * 16) * ldw + col0 + n;
    v16h b;
#pragma unroll
    for (int i = 0; i < 16; ++i) b[i] = (_Float16)p[(size_t)i * ldw];
    return b;
}

// Packed-B: fragment (kc, ct) stored lane-contiguously: 32 lanes x 16 f16.
__device__ __forceinline__ v16h load_b_packed(const _Float16* __restrict__ P,
                                              int CT, int kc, int ct) {
    const int lane = threadIdx.x & 31;
    return *(const v16h*)(P + (((size_t)kc * CT + ct) * 32 + lane) * 16);
}

// ---------------- weight pre-pack (f32 row-major -> f16 WMMA fragments) ----
// grid: (KC, CT, numMatrices); 32 threads
__global__ __launch_bounds__(32)
void pack_b_kernel(const float* __restrict__ W, _Float16* __restrict__ P,
                   int ldw, long long strW, long long strP)
{
    const int kc = blockIdx.x, ct = blockIdx.y, z = blockIdx.z;
    const float* Wz = W + (long long)z * strW;
    _Float16*    Pz = P + (long long)z * strP;
    const int lane = threadIdx.x & 31;
    v16h b = load_b_frag(Wz, ldw, kc * 32, ct * 16);
    _Float16* dst = Pz + (((size_t)kc * gridDim.y + ct) * 32 + lane) * 16;
    *(v16h*)dst = b;
}

// ---------------- strip GEMM: one wave = 16 rows x (NT*16) cols ------------
// out[t][row, :] = epilogue( A[t] @ W[t] + bias[t] ), weights pre-packed f16.
// C/D layout: acc[ct][i] -> row = row0 + (lane>>4)*8 + i, col = ct*16 + (lane&15)
template<int NT, int ACTIN, int ACTOUT, bool GATE>
__global__ __launch_bounds__(256)
void gemm_strip_kernel(const float* __restrict__ A, const _Float16* __restrict__ Pw,
                       const float* __restrict__ bias, float* __restrict__ out,
                       const float* __restrict__ resid, const float* __restrict__ skp,
                       int Mtiles, int Kdim, int ldA, int ldO,
                       long long strA, long long strPw, long long strB, long long strO)
{
    const int t = blockIdx.z;
    A    += (long long)t * strA;
    Pw   += (long long)t * strPw;
    bias += (long long)t * strB;
    out  += (long long)t * strO;
    if (GATE) resid += (long long)t * strO;

    const int mt = blockIdx.x * WPB_ + (threadIdx.x >> 5);
    if (mt >= Mtiles) return;                 // wave-uniform; EXEC stays full
    const int row0 = mt * 16;

    v8f acc[NT];
#pragma unroll
    for (int ct = 0; ct < NT; ++ct) acc[ct] = (v8f){};

    for (int k0 = 0; k0 < Kdim; k0 += 32) {
        const int kc = k0 >> 5;
        v16h a = load_a_frag<ACTIN>(A, ldA, row0, k0);
#pragma unroll
        for (int ct = 0; ct < NT; ++ct) {
            v16h b = load_b_packed(Pw, NT, kc, ct);
            acc[ct] = __builtin_amdgcn_wmma_f32_16x16x32_f16(
                false, a, false, b, (short)0, acc[ct], false, false);
        }
    }

    const int lane = threadIdx.x & 31;
    const int n  = lane & 15;
    const int hi = lane >> 4;
    float beta = 0.0f;
    if (GATE) beta = 1.0f / (1.0f + __expf(-skp[t]));

#pragma unroll
    for (int ct = 0; ct < NT; ++ct) {
        const int col = ct * 16 + n;
        const float bcol = bias[col];
#pragma unroll
        for (int i = 0; i < 8; ++i) {
            const int row = row0 + hi * 8 + i;
            float v = acc[ct][i] + bcol;
            if (ACTOUT == ACT_RELU) v = fmaxf(v, 0.0f);
            if (GATE) v = beta * v + (1.0f - beta) * resid[(size_t)row * ldO + col];
            out[(size_t)row * ldO + col] = v;
        }
    }
}

// ---------------- per-relation per-head D x D GEMM (K=32, 2 col tiles) -----
// out[r,n,hd,:] = KV[src_t[r], n, hd, :] @ REL[r,hd] ; REL pre-packed f16.
__global__ __launch_bounds__(256)
void rel_gemm_kernel(const float* __restrict__ KV, const _Float16* __restrict__ Prel,
                     const int* __restrict__ rel_src_type, float* __restrict__ outR)
{
    const int z  = blockIdx.z;             // r*HEADS + hd
    const int r  = z / HEADS_;
    const int hd = z % HEADS_;
    const int st = rel_src_type[r];

    const int mt = blockIdx.x * WPB_ + (threadIdx.x >> 5);
    if (mt >= MTILES_) return;
    const int row0 = mt * 16;

    const float*    A = KV   + (size_t)st * N_ * H_ + hd * D_;
    const _Float16* P = Prel + (size_t)z * (D_ * D_);     // 1024 f16 per (r,hd)
    float*        out = outR + (size_t)r  * N_ * H_ + hd * D_;

    v16h a = load_a_frag<ACT_NONE>(A, H_, row0, 0);
    v8f acc0 = (v8f){}, acc1 = (v8f){};
    acc0 = __builtin_amdgcn_wmma_f32_16x16x32_f16(
        false, a, false, load_b_packed(P, 2, 0, 0), (short)0, acc0, false, false);
    acc1 = __builtin_amdgcn_wmma_f32_16x16x32_f16(
        false, a, false, load_b_packed(P, 2, 0, 1), (short)0, acc1, false, false);

    const int lane = threadIdx.x & 31;
    const int n  = lane & 15;
    const int hi = lane >> 4;
#pragma unroll
    for (int i = 0; i < 8; ++i) {
        const int row = row0 + hi * 8 + i;
        out[(size_t)row * H_ + n]      = acc0[i];
        out[(size_t)row * H_ + 16 + n] = acc1[i];
    }
}

// ---------------- fill ----------------
__global__ void fill_kernel(float* __restrict__ p, float v, size_t n) {
    size_t i = (size_t)blockIdx.x * blockDim.x + threadIdx.x;
    if (i < n) p[i] = v;
}

// ---------------- edge pass 1: scores + running segment max ---------------
__global__ void edge_score_kernel(const float* __restrict__ q,
                                  const float* __restrict__ krel,
                                  const int* __restrict__ edge_src,
                                  const int* __restrict__ edge_dst,
                                  const int* __restrict__ rel_dst_type,
                                  const float* __restrict__ p_rel,
                                  float* __restrict__ sc,
                                  float* __restrict__ mmax)
{
    const int wave = (int)((blockIdx.x * blockDim.x + threadIdx.x) >> 5);
    const int lane = threadIdx.x & 31;
    if (wave >= R_ * E_) return;
    const int r = wave / E_;
    const int e = wave % E_;
    const int src = edge_src[r * E_ + e];
    const int dst = edge_dst[r * E_ + e];
    const int dt  = rel_dst_type[r];
    const float* qp = q    + ((size_t)dt * N_ + dst) * H_;
    const float* kp = krel + ((size_t)r  * N_ + src) * H_;
#pragma unroll
    for (int hd = 0; hd < HEADS_; ++hd) {
        float p = qp[hd * D_ + lane] * kp[hd * D_ + lane];
#pragma unroll
        for (int off = 16; off > 0; off >>= 1) p += __shfl_xor(p, off, 32);
        if (lane == 0) {
            const float s = p * p_rel[r * HEADS_ + hd] * 0.17677669529663687f; // 1/sqrt(D)
            sc[(size_t)wave * HEADS_ + hd] = s;
            atomicMaxFloat(&mmax[((size_t)r * N_ + dst) * HEADS_ + hd], s);
        }
    }
}

// ---------------- edge pass 2: exp + segment denom -------------------------
__global__ void edge_exp_kernel(const int* __restrict__ edge_dst,
                                const float* __restrict__ mmax,
                                float* __restrict__ sc,
                                float* __restrict__ denom)
{
    const int idx = blockIdx.x * blockDim.x + threadIdx.x;
    if (idx >= R_ * E_ * HEADS_) return;
    const int hd = idx % HEADS_;
    const int e  = (idx / HEADS_) % E_;
    const int r  = idx / (HEADS_ * E_);
    const int dst = edge_dst[r * E_ + e];
    const size_t seg = ((size_t)r * N_ + dst) * HEADS_ + hd;
    const float a = __expf(sc[idx] - mmax[seg]);
    sc[idx] = a;
    atomicAdd(&denom[seg], a);
}

// ---------------- edge pass 3: alpha-weighted scatter-add ------------------
__global__ void edge_agg_kernel(const float* __restrict__ vrel,
                                const int* __restrict__ edge_src,
                                const int* __restrict__ edge_dst,
                                const int* __restrict__ rel_dst_type,
                                const float* __restrict__ sc,
                                const float* __restrict__ denom,
                                float* __restrict__ agg)
{
    const int wave = (int)((blockIdx.x * blockDim.x + threadIdx.x) >> 5);
    const int lane = threadIdx.x & 31;
    if (wave >= R_ * E_) return;
    const int r = wave / E_;
    const int e = wave % E_;
    const int src = edge_src[r * E_ + e];
    const int dst = edge_dst[r * E_ + e];
    const int dt  = rel_dst_type[r];
    const float* vp = vrel + ((size_t)r * N_ + src) * H_;
    float* ap = agg + ((size_t)dt * N_ + dst) * H_;
    const size_t seg = ((size_t)r * N_ + dst) * HEADS_;
#pragma unroll
    for (int hd = 0; hd < HEADS_; ++hd) {
        const float alpha = sc[(size_t)wave * HEADS_ + hd] / denom[seg + hd];
        atomicAdd(&ap[hd * D_ + lane], vp[hd * D_ + lane] * alpha);
    }
}

// ===========================================================================
extern "C" void kernel_launch(void* const* d_in, const int* in_sizes, int n_in,
                              void* d_out, int out_size, void* d_ws, size_t ws_size,
                              hipStream_t stream)
{
    const float* x            = (const float*)d_in[0];
    const int*   edge_src     = (const int*)  d_in[1];
    const int*   edge_dst     = (const int*)  d_in[2];
    const int*   rel_src_type = (const int*)  d_in[3];
    const int*   rel_dst_type = (const int*)  d_in[4];
    const float* Wenc = (const float*)d_in[5];
    const float* benc = (const float*)d_in[6];
    const float* Wk   = (const float*)d_in[7];
    const float* bk   = (const float*)d_in[8];
    const float* Wq   = (const float*)d_in[9];
    const float* bq   = (const float*)d_in[10];
    const float* Wv   = (const float*)d_in[11];
    const float* bv   = (const float*)d_in[12];
    const float* Wa   = (const float*)d_in[13];
    const float* ba   = (const float*)d_in[14];
    const float* skip = (const float*)d_in[15];
    const float* a_rel = (const float*)d_in[16];
    const float* m_rel = (const float*)d_in[17];
    const float* p_rel = (const float*)d_in[18];
    const float* Wout = (const float*)d_in[19];
    const float* bout = (const float*)d_in[20];
    float* out = (float*)d_out;

    // ---- workspace layout ----
    const size_t TNH = (size_t)T_ * N_ * H_;     // 20.48 M floats
    const size_t RNH = (size_t)R_ * N_ * H_;     // 30.72 M floats
    float* ws   = (float*)d_ws;
    float* h0   = ws;
    float* h1   = h0 + TNH;
    float* kb   = h1 + TNH;          // k buffer; reused as agg after k_rel formed
    float* qb   = kb + TNH;
    float* vb   = qb + TNH;
    float* krel = vb + TNH;          // (R,N,H)
    float* vrel = krel + RNH;        // (R,N,H)
    float* sc   = vrel + RNH;                         // (R*E, HEADS)
    float* mm   = sc + (size_t)R_ * E_ * HEADS_;      // (R*N, HEADS)
    float* dn   = mm + (size_t)R_ * N_ * HEADS_;      // (R*N, HEADS)

    // packed f16 weights at tail (32B-aligned: all offsets multiple of 8 floats)
    _Float16* pk = (_Float16*)(dn + (size_t)R_ * N_ * HEADS_);
    const size_t MAT  = (size_t)H_ * H_;         // 16384 f16 per packed 128x128
    const size_t RELM = (size_t)R_ * HEADS_ * D_ * D_;  // 49152 f16 per rel tensor
    _Float16* pWenc = pk;                        // T_*MAT
    _Float16* pWk   = pWenc + T_ * MAT;          // L_*T_*MAT each below
    _Float16* pWq   = pWk + (size_t)L_ * T_ * MAT;
    _Float16* pWv   = pWq + (size_t)L_ * T_ * MAT;
    _Float16* pWa   = pWv + (size_t)L_ * T_ * MAT;
    _Float16* pWout = pWa + (size_t)L_ * T_ * MAT;      // 2048
    _Float16* pArel = pWout + (size_t)FIN_ * C_;        // L_*RELM
    _Float16* pMrel = pArel + (size_t)L_ * RELM;        // L_*RELM

    const dim3 blk32(32);
    const dim3 blk256(256);

    // ---- pre-pack all weights into WMMA f16 fragment layout ----
    {
        const dim3 gFullW(4, 8, T_);             // KC=4, CT=8, per type
        pack_b_kernel<<<gFullW, blk32, 0, stream>>>(Wenc, pWenc, H_, (long long)FIN_ * H_, (long long)MAT);
        const dim3 gLTW(4, 8, L_ * T_);          // all layers+types at once
        pack_b_kernel<<<gLTW, blk32, 0, stream>>>(Wk, pWk, H_, (long long)H_ * H_, (long long)MAT);
        pack_b_kernel<<<gLTW, blk32, 0, stream>>>(Wq, pWq, H_, (long long)H_ * H_, (long long)MAT);
        pack_b_kernel<<<gLTW, blk32, 0, stream>>>(Wv, pWv, H_, (long long)H_ * H_, (long long)MAT);
        pack_b_kernel<<<gLTW, blk32, 0, stream>>>(Wa, pWa, H_, (long long)H_ * H_, (long long)MAT);
        const dim3 gOutW(4, 1, 1);               // KC=4, CT=1
        pack_b_kernel<<<gOutW, blk32, 0, stream>>>(Wout, pWout, C_, 0LL, 0LL);
        const dim3 gRelW(1, 2, L_ * R_ * HEADS_); // KC=1, CT=2, per (l,r,hd)
        pack_b_kernel<<<gRelW, blk32, 0, stream>>>(a_rel, pArel, D_, (long long)D_ * D_, (long long)D_ * D_);
        pack_b_kernel<<<gRelW, blk32, 0, stream>>>(m_rel, pMrel, D_, (long long)D_ * D_, (long long)D_ * D_);
    }

    const dim3 gStrip(MBLKS_, 1, T_);            // (157, 1, 8)
    const dim3 gRelG (MBLKS_, 1, R_ * HEADS_);   // (157, 1, 48)
    const long long sNH = (long long)N_ * H_;
    const long long sM  = (long long)MAT;

    // ---- encoder: h0 = relu(x @ Wenc + benc) ----
    gemm_strip_kernel<8, ACT_NONE, ACT_RELU, false><<<gStrip, blk256, 0, stream>>>(
        x, pWenc, benc, h0, nullptr, nullptr,
        MTILES_, FIN_, FIN_, H_, (long long)N_ * FIN_, sM, (long long)H_, sNH);

    float* hcur = h0;
    float* hnext = h1;

    const int edgeWaves  = R_ * E_;                                  // 480000
    const int edgeBlocks = (edgeWaves * 32 + 255) / 256;
    const int ehBlocks   = (R_ * E_ * HEADS_ + 255) / 256;

    for (int l = 0; l < L_; ++l) {
        const _Float16* pWk_l = pWk + (size_t)l * T_ * MAT;
        const _Float16* pWq_l = pWq + (size_t)l * T_ * MAT;
        const _Float16* pWv_l = pWv + (size_t)l * T_ * MAT;
        const _Float16* pWa_l = pWa + (size_t)l * T_ * MAT;
        const _Float16* pArel_l = pArel + (size_t)l * RELM;
        const _Float16* pMrel_l = pMrel + (size_t)l * RELM;
        const float* bk_l = bk + (size_t)l * T_ * H_;
        const float* bq_l = bq + (size_t)l * T_ * H_;
        const float* bv_l = bv + (size_t)l * T_ * H_;
        const float* ba_l = ba + (size_t)l * T_ * H_;
        const float* p_rel_l = p_rel + (size_t)l * R_ * HEADS_;
        const float* skip_l  = skip  + (size_t)l * T_;

        // k, q, v projections
        gemm_strip_kernel<8, ACT_NONE, ACT_NONE, false><<<gStrip, blk256, 0, stream>>>(
            hcur, pWk_l, bk_l, kb, nullptr, nullptr,
            MTILES_, H_, H_, H_, sNH, sM, (long long)H_, sNH);
        gemm_strip_kernel<8, ACT_NONE, ACT_NONE, false><<<gStrip, blk256, 0, stream>>>(
            hcur, pWq_l, bq_l, qb, nullptr, nullptr,
            MTILES_, H_, H_, H_, sNH, sM, (long long)H_, sNH);
        gemm_strip_kernel<8, ACT_NONE, ACT_NONE, false><<<gStrip, blk256, 0, stream>>>(
            hcur, pWv_l, bv_l, vb, nullptr, nullptr,
            MTILES_, H_, H_, H_, sNH, sM, (long long)H_, sNH);

        // per-relation per-head rotations: k_rel, v_rel
        rel_gemm_kernel<<<gRelG, blk256, 0, stream>>>(kb, pArel_l, rel_src_type, krel);
        rel_gemm_kernel<<<gRelG, blk256, 0, stream>>>(vb, pMrel_l, rel_src_type, vrel);

        // init softmax state + agg (kb is dead now -> reuse as agg)
        {
            const size_t nSeg = (size_t)R_ * N_ * HEADS_;
            fill_kernel<<<(int)((nSeg + 255) / 256), 256, 0, stream>>>(mm, -1e30f, nSeg);
            fill_kernel<<<(int)((nSeg + 255) / 256), 256, 0, stream>>>(dn, 0.0f, nSeg);
            fill_kernel<<<(int)((TNH + 255) / 256), 256, 0, stream>>>(kb, 0.0f, TNH);
        }

        // edge phase: scores+max, exp+denom, alpha*v scatter
        edge_score_kernel<<<edgeBlocks, 256, 0, stream>>>(
            qb, krel, edge_src, edge_dst, rel_dst_type, p_rel_l, sc, mm);
        edge_exp_kernel<<<ehBlocks, 256, 0, stream>>>(edge_dst, mm, sc, dn);
        edge_agg_kernel<<<edgeBlocks, 256, 0, stream>>>(
            vrel, edge_src, edge_dst, rel_dst_type, sc, dn, kb /*agg*/);

        // out = gelu(agg) @ Wa + ba ; h_next = beta*out + (1-beta)*h
        gemm_strip_kernel<8, ACT_GELU, ACT_NONE, true><<<gStrip, blk256, 0, stream>>>(
            kb, pWa_l, ba_l, hnext, hcur, skip_l,
            MTILES_, H_, H_, H_, sNH, sM, (long long)H_, sNH);

        float* tmp = hcur; hcur = hnext; hnext = tmp;
    }

    // final head: out = h[0] @ Wout + bout  (C_=16 -> NT=1)
    const dim3 gFin(MBLKS_, 1, 1);
    gemm_strip_kernel<1, ACT_NONE, ACT_NONE, false><<<gFin, blk256, 0, stream>>>(
        hcur, pWout, bout, out, nullptr, nullptr,
        MTILES_, H_, H_, C_, 0LL, 0LL, 0LL, 0LL);
}
